// TrippleAttention_20452634263749
// MI455X (gfx1250) — compile-verified, matrix operand order checked by reading, not run on hardware
//
#include <hip/hip_runtime.h>
#include <hip/hip_bf16.h>

typedef __attribute__((ext_vector_type(16))) __bf16 v16bf;
typedef __attribute__((ext_vector_type(8)))  float  v8f;

// ---- problem constants ----
#define BB    4
#define CC    60
#define NANG  25
#define HH    96
#define WWID  96
#define HW    9216
#define H2W   2304          // 48*48
#define PTOT  921600L       // B*N*H*W
#define PLOW  230400L       // B*N*48*48
#define DNDIM 1500          // d*n
#define NHWB  230400        // N*H*W per batch

// ---------- WMMA helpers (wave32, 16x16x32 bf16 -> f32) ----------
__device__ __forceinline__ int frag_k(int e, int hf) {
    int j = e >> 1, lo = e & 1;
    return ((j >> 2) << 4) + hf * 8 + ((j & 3) << 1) + lo;
}

// Generic fragment loader. Element (rc, k) lives at p[rc*rc_stride + k*k_stride].
// A-fragment: rc = row(M).  B-fragment: rc = col(N).
__device__ __forceinline__ v16bf load_frag(const __bf16* p, int rc_stride, int k_stride,
                                           int rc_base, int lane) {
    int hf = lane >> 4;
    int rc = rc_base + (lane & 15);
    v16bf a;
#pragma unroll
    for (int e = 0; e < 16; ++e)
        a[e] = p[(long)rc * rc_stride + (long)frag_k(e, hf) * k_stride];
    return a;
}

__device__ __forceinline__ v8f wmma_bf16(v16bf a, v16bf b, v8f c) {
    return __builtin_amdgcn_wmma_f32_16x16x32_bf16(false, a, false, b, (short)0, c, false, false);
}

// ====================================================================
// K1: fused full-res projections: [w_spa_q;w_ang_qkv;w_fre_qkv] (420x60) @ x
//     outputs channel-major bf16: PQ[60][P], PA[180][P], PF[180][P]
// grid 14400, block 128 (4 waves); tile: 432 oc x 64 positions, K=64(pad)
// ====================================================================
__global__ void k_proj_full(const float* __restrict__ x,
                            const float* __restrict__ wq,
                            const float* __restrict__ wa,
                            const float* __restrict__ wf,
                            __bf16* __restrict__ PQ, __bf16* __restrict__ PA,
                            __bf16* __restrict__ PF) {
    __shared__ __bf16 sw[432 * 64];   // [row][k]
    __shared__ __bf16 sx[64 * 64];    // [k][pos]
    int t = threadIdx.x;
    for (int idx = t; idx < 432 * 64; idx += 128) {
        int r = idx >> 6, k = idx & 63;
        float v = 0.f;
        if (k < 60) {
            if (r < 60)       v = wq[r * 60 + k];
            else if (r < 240) v = wa[(r - 60) * 60 + k];
            else if (r < 420) v = wf[(r - 240) * 60 + k];
        }
        sw[idx] = (__bf16)v;
    }
    long p0 = (long)blockIdx.x * 64;
    int b = (int)(p0 / (NANG * HW));
    int nhw = (int)(p0 % (NANG * HW));
    for (int idx = t; idx < 64 * 64; idx += 128) {
        int k = idx >> 6, i = idx & 63;
        float v = 0.f;
        if (k < 60) v = x[((long)b * CC + k) * (long)(NANG * HW) + nhw + i];
        sx[idx] = (__bf16)v;
    }
    __syncthreads();
    int wave = t >> 5, lane = t & 31;
    int ci = wave * 16;
    v16bf b0 = load_frag(sx,            1, 64, ci, lane);
    v16bf b1 = load_frag(sx + 32 * 64,  1, 64, ci, lane);
    int coln = lane & 15, hf = lane >> 4;
    for (int mt = 0; mt < 27; ++mt) {
        v8f acc = {};
        v16bf a0 = load_frag(sw,      64, 1, mt * 16, lane);
        v16bf a1 = load_frag(sw + 32, 64, 1, mt * 16, lane);
        acc = wmma_bf16(a0, b0, acc);
        acc = wmma_bf16(a1, b1, acc);
        long pp = p0 + ci + coln;
#pragma unroll
        for (int v = 0; v < 8; ++v) {
            int oc = mt * 16 + v + hf * 8;
            float val = acc[v];
            if (oc < 60)       PQ[(long)oc * PTOT + pp]        = (__bf16)val;
            else if (oc < 240) PA[(long)(oc - 60) * PTOT + pp]  = (__bf16)val;
            else if (oc < 420) PF[(long)(oc - 240) * PTOT + pp] = (__bf16)val;
        }
    }
}

// ====================================================================
// K2: low-res projection spa_kv (120x60) @ x[:,:,:,::2,::2] -> PKV[120][PL]
// grid 3600, block 128
// ====================================================================
__global__ void k_proj_low(const float* __restrict__ x, const float* __restrict__ wkv,
                           __bf16* __restrict__ PKV) {
    __shared__ __bf16 sw[128 * 64];
    __shared__ __bf16 sx[64 * 64];
    int t = threadIdx.x;
    for (int idx = t; idx < 128 * 64; idx += 128) {
        int r = idx >> 6, k = idx & 63;
        float v = (r < 120 && k < 60) ? wkv[r * 60 + k] : 0.f;
        sw[idx] = (__bf16)v;
    }
    long pl0 = (long)blockIdx.x * 64;
    for (int idx = t; idx < 64 * 64; idx += 128) {
        int k = idx >> 6, i = idx & 63;
        float v = 0.f;
        if (k < 60) {
            long pl = pl0 + i;
            int b  = (int)(pl / (NANG * H2W));
            int r  = (int)(pl % (NANG * H2W));
            int n  = r / H2W;  int rr = r % H2W;
            int h2 = rr / 48,  w2 = rr % 48;
            v = x[(((long)b * CC + k) * NANG + n) * HW + (2 * h2) * WWID + 2 * w2];
        }
        sx[idx] = (__bf16)v;
    }
    __syncthreads();
    int wave = t >> 5, lane = t & 31;
    int ci = wave * 16;
    v16bf b0 = load_frag(sx,           1, 64, ci, lane);
    v16bf b1 = load_frag(sx + 32 * 64, 1, 64, ci, lane);
    int coln = lane & 15, hf = lane >> 4;
    for (int mt = 0; mt < 8; ++mt) {
        v8f acc = {};
        v16bf a0 = load_frag(sw,      64, 1, mt * 16, lane);
        v16bf a1 = load_frag(sw + 32, 64, 1, mt * 16, lane);
        acc = wmma_bf16(a0, b0, acc);
        acc = wmma_bf16(a1, b1, acc);
#pragma unroll
        for (int v = 0; v < 8; ++v) {
            int oc = mt * 16 + v + hf * 8;
            if (oc < 120) PKV[(long)oc * PLOW + pl0 + ci + coln] = (__bf16)acc[v];
        }
    }
}

// ====================================================================
// K3/K4: spatial norm scales. rq over (d,n) per (b,hw); rk per (b,h2w2).
// ====================================================================
__global__ void k_stat_sq(const __bf16* __restrict__ PQ, float* __restrict__ rq) {
    int g = blockIdx.x * 256 + threadIdx.x;
    if (g >= BB * HW) return;
    int b = g / HW, i = g % HW;
    float s = 0.f;
    for (int d = 0; d < 60; ++d)
        for (int n = 0; n < NANG; ++n) {
            float v = (float)PQ[(long)d * PTOT + ((long)(b * NANG + n)) * HW + i];
            s += v * v;
        }
    rq[g] = 1.f / fmaxf(sqrtf(s), 1e-12f);
}

__global__ void k_stat_sk(const __bf16* __restrict__ PKV, float* __restrict__ rk) {
    int g = blockIdx.x * 256 + threadIdx.x;
    if (g >= BB * H2W) return;
    int b = g / H2W, j = g % H2W;
    float s = 0.f;
    for (int d = 0; d < 60; ++d)
        for (int n = 0; n < NANG; ++n) {
            float v = (float)PKV[(long)d * PLOW + ((long)(b * NANG + n)) * H2W + j];
            s += v * v;
        }
    rk[g] = 1.f / fmaxf(sqrtf(s), 1e-12f);
}

// ====================================================================
// K5: angular norm scales per (b,n) over (d,hw). grid 100, block 256.
// ====================================================================
__global__ void k_stat_ang(const __bf16* __restrict__ PA, float* __restrict__ rqa,
                           float* __restrict__ rka) {
    int bn = blockIdx.x;
    int b = bn / NANG, n = bn % NANG;
    long base = ((long)(b * NANG + n)) * HW;
    float sq = 0.f, sk = 0.f;
    for (int d = 0; d < 60; ++d) {
        const __bf16* q = PA + (long)d * PTOT + base;
        const __bf16* k = PA + (long)(60 + d) * PTOT + base;
        for (int i = threadIdx.x; i < HW; i += 256) {
            float a = (float)q[i]; sq += a * a;
            float c = (float)k[i]; sk += c * c;
        }
    }
    __shared__ float r1[256], r2[256];
    r1[threadIdx.x] = sq; r2[threadIdx.x] = sk; __syncthreads();
    for (int s = 128; s > 0; s >>= 1) {
        if (threadIdx.x < s) { r1[threadIdx.x] += r1[threadIdx.x + s]; r2[threadIdx.x] += r2[threadIdx.x + s]; }
        __syncthreads();
    }
    if (threadIdx.x == 0) {
        rqa[bn] = 1.f / fmaxf(sqrtf(r1[0]), 1e-12f);
        rka[bn] = 1.f / fmaxf(sqrtf(r2[0]), 1e-12f);
    }
}

// ====================================================================
// K6: frequency norm scales per (b,d) over (n,hw). grid 240, block 256.
// ====================================================================
__global__ void k_stat_fre(const __bf16* __restrict__ PF, float* __restrict__ rqf,
                           float* __restrict__ rkf) {
    int bd = blockIdx.x;
    int b = bd / 60, d = bd % 60;
    long base = (long)b * NHWB;
    const __bf16* q = PF + (long)d * PTOT + base;
    const __bf16* k = PF + (long)(60 + d) * PTOT + base;
    float sq = 0.f, sk = 0.f;
    for (int i = threadIdx.x; i < NHWB; i += 256) {
        float a = (float)q[i]; sq += a * a;
        float c = (float)k[i]; sk += c * c;
    }
    __shared__ float r1[256], r2[256];
    r1[threadIdx.x] = sq; r2[threadIdx.x] = sk; __syncthreads();
    for (int s = 128; s > 0; s >>= 1) {
        if (threadIdx.x < s) { r1[threadIdx.x] += r1[threadIdx.x + s]; r2[threadIdx.x] += r2[threadIdx.x + s]; }
        __syncthreads();
    }
    if (threadIdx.x == 0) {
        rqf[bd] = 1.f / fmaxf(sqrtf(r1[0]), 1e-12f);
        rkf[bd] = 1.f / fmaxf(sqrtf(r2[0]), 1e-12f);
    }
}

// ====================================================================
// K7: spatial QK raw scores  S[b][i][j] = sum_dn Q[dn,i]K[dn,j]
// grid (36 jt, 144 it, 4 b), block 256, tile 64x64, K=1504 pad (47 steps)
// ====================================================================
__global__ void k_qk_s(const __bf16* __restrict__ PQ, const __bf16* __restrict__ PKV,
                       __bf16* __restrict__ S) {
    __shared__ __bf16 sA[32 * 64], sB[32 * 64];   // [k][i] / [k][j]
    int b = blockIdx.z;
    int i0 = blockIdx.y * 64, j0 = blockIdx.x * 64;
    int t = threadIdx.x, wave = t >> 5, lane = t & 31;
    int mt = wave & 3, nt = (wave >> 2) * 2;
    v8f acc0 = {}, acc1 = {};
    for (int ks = 0; ks < 47; ++ks) {
        __syncthreads();
        for (int idx = t; idx < 2048; idx += 256) {
            int kk = idx >> 6, c = idx & 63;
            int dn = ks * 32 + kk;
            float va = 0.f, vb = 0.f;
            if (dn < DNDIM) {
                int d = dn / NANG, n = dn % NANG;
                long rw = (long)(b * NANG + n);
                const __bf16* pa = &PQ[(long)d * PTOT + rw * HW + i0 + c];
                const __bf16* pb = &PKV[(long)d * PLOW + rw * H2W + j0 + c];
                va = (float)*pa;  vb = (float)*pb;
                int dn2 = dn + 32;
                if (dn2 < DNDIM) {      // CDNA5 global_prefetch of next K-tile
                    int d2 = dn2 / NANG, n2 = dn2 % NANG;
                    long rw2 = (long)(b * NANG + n2);
                    __builtin_prefetch(&PQ[(long)d2 * PTOT + rw2 * HW + i0 + c], 0, 0);
                    __builtin_prefetch(&PKV[(long)d2 * PLOW + rw2 * H2W + j0 + c], 0, 0);
                }
            }
            sA[idx] = (__bf16)va;  sB[idx] = (__bf16)vb;
        }
        __syncthreads();
        v16bf a  = load_frag(sA, 1, 64, mt * 16, lane);
        v16bf f0 = load_frag(sB, 1, 64, nt * 16, lane);
        v16bf f1 = load_frag(sB, 1, 64, nt * 16 + 16, lane);
        acc0 = wmma_bf16(a, f0, acc0);
        acc1 = wmma_bf16(a, f1, acc1);
    }
    int coln = lane & 15, hf = lane >> 4;
    long rb = (long)b * HW * H2W;
#pragma unroll
    for (int v = 0; v < 8; ++v) {
        int i = i0 + mt * 16 + v + hf * 8;
        long off = rb + (long)i * H2W + j0;
        S[off + nt * 16 + coln]      = (__bf16)acc0[v];
        S[off + nt * 16 + 16 + coln] = (__bf16)acc1[v];
    }
}

// ====================================================================
// K8: spatial softmax in place. grid 36864, block 256 (9 cols/thread).
// ====================================================================
__global__ void k_softmax_s(__bf16* __restrict__ S, const float* __restrict__ rq,
                            const float* __restrict__ rk) {
    int g = blockIdx.x;
    int b = g / HW;
    __bf16* row = S + (long)g * H2W;
    float qs = rq[g];
    __shared__ float red[256];
    float vals[9];
    float mx = -1e30f;
#pragma unroll
    for (int r = 0; r < 9; ++r) {
        int j = threadIdx.x + r * 256;
        float v = (float)row[j] * qs * rk[b * H2W + j];
        vals[r] = v; mx = fmaxf(mx, v);
    }
    red[threadIdx.x] = mx; __syncthreads();
    for (int s = 128; s > 0; s >>= 1) {
        if (threadIdx.x < s) red[threadIdx.x] = fmaxf(red[threadIdx.x], red[threadIdx.x + s]);
        __syncthreads();
    }
    mx = red[0]; __syncthreads();
    float sum = 0.f;
#pragma unroll
    for (int r = 0; r < 9; ++r) { vals[r] = __expf(vals[r] - mx); sum += vals[r]; }
    red[threadIdx.x] = sum; __syncthreads();
    for (int s = 128; s > 0; s >>= 1) {
        if (threadIdx.x < s) red[threadIdx.x] += red[threadIdx.x + s];
        __syncthreads();
    }
    float inv = 1.f / red[0];
#pragma unroll
    for (int r = 0; r < 9; ++r) row[threadIdx.x + r * 256] = (__bf16)(vals[r] * inv);
}

// ====================================================================
// K9: angular gram (25x25 pad 32) over K=552960; K-split + f32 atomics.
// grid (270, 4), block 128 (4 waves, one 16x16 tile each), 64 K-steps.
// ====================================================================
__global__ void k_gram_a(const __bf16* __restrict__ PA, float* __restrict__ Ga) {
    __shared__ __bf16 sA[32 * 32], sB[32 * 32];   // [k][n]
    int b = blockIdx.y;
    long k0 = (long)blockIdx.x * 2048;
    int t = threadIdx.x, wave = t >> 5, lane = t & 31;
    int mt = wave >> 1, nt = wave & 1;
    v8f acc = {};
    for (int ks = 0; ks < 64; ++ks) {
        long kb = k0 + (long)ks * 32;
        int d = (int)(kb / HW); int hw = (int)(kb % HW);
        __syncthreads();
        for (int idx = t; idx < 1024; idx += 128) {
            int kk = idx & 31, n = idx >> 5;
            float va = 0.f, vb = 0.f;
            if (n < NANG) {
                long p = ((long)(b * NANG + n)) * HW + hw + kk;
                va = (float)PA[(long)d * PTOT + p];
                vb = (float)PA[(long)(60 + d) * PTOT + p];
            }
            sA[kk * 32 + n] = (__bf16)va;
            sB[kk * 32 + n] = (__bf16)vb;
        }
        __syncthreads();
        v16bf a = load_frag(sA, 1, 32, mt * 16, lane);
        v16bf f = load_frag(sB, 1, 32, nt * 16, lane);
        acc = wmma_bf16(a, f, acc);
    }
    int coln = lane & 15, hf = lane >> 4;
#pragma unroll
    for (int v = 0; v < 8; ++v) {
        int m = mt * 16 + v + hf * 8;
        atomicAdd(&Ga[(b * 32 + m) * 32 + nt * 16 + coln], acc[v]);
    }
}

// ====================================================================
// K10: frequency gram (60x60 pad 64) over K=230400. grid (225,4), block 256.
// ====================================================================
__global__ void k_gram_f(const __bf16* __restrict__ PF, float* __restrict__ Gf) {
    __shared__ __bf16 sA[32 * 64], sB[32 * 64];   // [k][d]
    int b = blockIdx.y;
    long k0 = (long)blockIdx.x * 1024;
    int t = threadIdx.x, wave = t >> 5, lane = t & 31;
    int mt = wave & 3, nt = (wave >> 2) * 2;
    long base = (long)b * NHWB;
    v8f acc0 = {}, acc1 = {};
    for (int ks = 0; ks < 32; ++ks) {
        long kb = k0 + (long)ks * 32;
        __syncthreads();
        for (int idx = t; idx < 2048; idx += 256) {
            int kk = idx & 31, dch = idx >> 5;
            float va = 0.f, vb = 0.f;
            if (dch < 60) {
                va = (float)PF[(long)dch * PTOT + base + kb + kk];
                vb = (float)PF[(long)(60 + dch) * PTOT + base + kb + kk];
            }
            sA[kk * 64 + dch] = (__bf16)va;
            sB[kk * 64 + dch] = (__bf16)vb;
        }
        __syncthreads();
        v16bf a  = load_frag(sA, 1, 64, mt * 16, lane);
        v16bf f0 = load_frag(sB, 1, 64, nt * 16, lane);
        v16bf f1 = load_frag(sB, 1, 64, nt * 16 + 16, lane);
        acc0 = wmma_bf16(a, f0, acc0);
        acc1 = wmma_bf16(a, f1, acc1);
    }
    int coln = lane & 15, hf = lane >> 4;
#pragma unroll
    for (int v = 0; v < 8; ++v) {
        int m = mt * 16 + v + hf * 8;
        atomicAdd(&Gf[(b * 64 + m) * 64 + nt * 16 + coln], acc0[v]);
        atomicAdd(&Gf[(b * 64 + m) * 64 + nt * 16 + 16 + coln], acc1[v]);
    }
}

// ====================================================================
// K11/K12: tiny softmaxes with norm-scale folding; zero padded rows/cols.
// ====================================================================
__global__ void k_softmax_a(const float* __restrict__ Ga, const float* __restrict__ rqa,
                            const float* __restrict__ rka, __bf16* __restrict__ Aatt) {
    int g = threadIdx.x;                  // 128 = 4b * 32 rows
    int b = g >> 5, m = g & 31;
    float mx = -1e30f;
    for (int j = 0; j < NANG; ++j)
        if (m < NANG) mx = fmaxf(mx, Ga[(b * 32 + m) * 32 + j] * rqa[b * NANG + m] * rka[b * NANG + j]);
    float s = 0.f;
    for (int j = 0; j < NANG; ++j)
        if (m < NANG) s += __expf(Ga[(b * 32 + m) * 32 + j] * rqa[b * NANG + m] * rka[b * NANG + j] - mx);
    float inv = (m < NANG) ? 1.f / s : 0.f;
    for (int j = 0; j < 32; ++j) {
        float e = 0.f;
        if (m < NANG && j < NANG)
            e = __expf(Ga[(b * 32 + m) * 32 + j] * rqa[b * NANG + m] * rka[b * NANG + j] - mx) * inv;
        Aatt[(b * 32 + m) * 32 + j] = (__bf16)e;
    }
}

__global__ void k_softmax_f(const float* __restrict__ Gf, const float* __restrict__ rqf,
                            const float* __restrict__ rkf, __bf16* __restrict__ Fatt) {
    int g = threadIdx.x;                  // 256 = 4b * 64 rows
    int b = g >> 6, m = g & 63;
    float mx = -1e30f;
    for (int j = 0; j < 60; ++j)
        if (m < 60) mx = fmaxf(mx, Gf[(b * 64 + m) * 64 + j] * rqf[b * 60 + m] * rkf[b * 60 + j]);
    float s = 0.f;
    for (int j = 0; j < 60; ++j)
        if (m < 60) s += __expf(Gf[(b * 64 + m) * 64 + j] * rqf[b * 60 + m] * rkf[b * 60 + j] - mx);
    float inv = (m < 60) ? 1.f / s : 0.f;
    for (int j = 0; j < 64; ++j) {
        float e = 0.f;
        if (m < 60 && j < 60)
            e = __expf(Gf[(b * 64 + m) * 64 + j] * rqf[b * 60 + m] * rkf[b * 60 + j] - mx) * inv;
        Fatt[(b * 64 + m) * 64 + j] = (__bf16)e;
    }
}

// ====================================================================
// K13: spatial PV: SF[b][i][dn] = P @ sv. grid (24 dn-t, 144 i-t, 4 b), K=2304.
// ====================================================================
__global__ void k_pv_s(const __bf16* __restrict__ S, const __bf16* __restrict__ PKV,
                       __bf16* __restrict__ SF) {
    __shared__ __bf16 sA[64 * 32], sB[64 * 32];   // [i][k] / [dn][k]
    int b = blockIdx.z;
    int i0 = blockIdx.y * 64, c0 = blockIdx.x * 64;
    int t = threadIdx.x, wave = t >> 5, lane = t & 31;
    int mt = wave & 3, nt = (wave >> 2) * 2;
    const __bf16* Sb = S + (long)b * HW * H2W;
    v8f acc0 = {}, acc1 = {};
    for (int ks = 0; ks < 72; ++ks) {
        int j0 = ks * 32;
        __syncthreads();
        for (int idx = t; idx < 2048; idx += 256) {
            int kk = idx & 31, r = idx >> 5;
            sA[r * 32 + kk] = Sb[(long)(i0 + r) * H2W + j0 + kk];
            int dn = c0 + r;
            float vb = 0.f;
            if (dn < DNDIM) {
                int d = dn / NANG, n = dn % NANG;
                vb = (float)PKV[(long)(60 + d) * PLOW + ((long)(b * NANG + n)) * H2W + j0 + kk];
            }
            sB[r * 32 + kk] = (__bf16)vb;
        }
        __syncthreads();
        v16bf a  = load_frag(sA, 32, 1, mt * 16, lane);
        v16bf f0 = load_frag(sB, 32, 1, nt * 16, lane);
        v16bf f1 = load_frag(sB, 32, 1, (nt + 1) * 16, lane);
        acc0 = wmma_bf16(a, f0, acc0);
        acc1 = wmma_bf16(a, f1, acc1);
    }
    int coln = lane & 15, hf = lane >> 4;
    long sfb = (long)b * HW * DNDIM;
#pragma unroll
    for (int v = 0; v < 8; ++v) {
        int i = i0 + mt * 16 + v + hf * 8;
        int c1 = c0 + nt * 16 + coln, c2 = c0 + (nt + 1) * 16 + coln;
        if (c1 < DNDIM) SF[sfb + (long)i * DNDIM + c1] = (__bf16)acc0[v];
        if (c2 < DNDIM) SF[sfb + (long)i * DNDIM + c2] = (__bf16)acc1[v];
    }
}

// ====================================================================
// K14: angular PV: AF[d][p] = Aatt @ av. grid (8640, 4), block 256, K=32.
// ====================================================================
__global__ void k_pv_a(const __bf16* __restrict__ Aatt, const __bf16* __restrict__ PA,
                       __bf16* __restrict__ AF) {
    __shared__ __bf16 sB[32 * 64];   // [n2][col]
    int b = blockIdx.y;
    long c0 = (long)blockIdx.x * 64;     // column in (d,hw)
    int d = (int)(c0 / HW); int hw0 = (int)(c0 % HW);
    int t = threadIdx.x, wave = t >> 5, lane = t & 31;
    for (int idx = t; idx < 2048; idx += 256) {
        int n2 = idx >> 6, c = idx & 63;
        float v = 0.f;
        if (n2 < NANG)
            v = (float)PA[(long)(120 + d) * PTOT + ((long)(b * NANG + n2)) * HW + hw0 + c];
        sB[n2 * 64 + c] = (__bf16)v;
    }
    __syncthreads();
    int mt = wave & 1, nt = wave >> 1;
    v16bf a = load_frag(Aatt + (long)b * 1024, 32, 1, mt * 16, lane);
    v16bf f = load_frag(sB, 1, 64, nt * 16, lane);
    v8f acc = {};
    acc = wmma_bf16(a, f, acc);
    int coln = lane & 15, hf = lane >> 4;
#pragma unroll
    for (int v = 0; v < 8; ++v) {
        int m = mt * 16 + v + hf * 8;   // n1
        if (m < NANG)
            AF[(long)d * PTOT + ((long)(b * NANG + m)) * HW + hw0 + nt * 16 + coln] = (__bf16)acc[v];
    }
}

// ====================================================================
// K15: frequency PV: FF[d][p] = Fatt @ fv. grid (7200, 4), block 256, K=64.
// ====================================================================
__global__ void k_pv_f(const __bf16* __restrict__ Fatt, const __bf16* __restrict__ PF,
                       __bf16* __restrict__ FF) {
    __shared__ __bf16 sB[64 * 32];   // [d2(k)][col]
    int b = blockIdx.y;
    long c0 = (long)blockIdx.x * 32;     // column in (n,hw) == p within b
    int t = threadIdx.x, wave = t >> 5, lane = t & 31;
    for (int idx = t; idx < 2048; idx += 256) {
        int d2 = idx >> 5, c = idx & 31;
        float v = 0.f;
        if (d2 < 60) v = (float)PF[(long)(120 + d2) * PTOT + (long)b * NHWB + c0 + c];
        sB[d2 * 32 + c] = (__bf16)v;
    }
    __syncthreads();
    int mt = wave & 3, nt = wave >> 2;
    const __bf16* A = Fatt + (long)b * 4096;      // [64][64] row-major (m,k)
    v16bf a0 = load_frag(A,      64, 1, mt * 16, lane);
    v16bf a1 = load_frag(A + 32, 64, 1, mt * 16, lane);
    v16bf f0 = load_frag(sB,           1, 32, nt * 16, lane);
    v16bf f1 = load_frag(sB + 32 * 32, 1, 32, nt * 16, lane);
    v8f acc = {};
    acc = wmma_bf16(a0, f0, acc);
    acc = wmma_bf16(a1, f1, acc);
    int coln = lane & 15, hf = lane >> 4;
#pragma unroll
    for (int v = 0; v < 8; ++v) {
        int m = mt * 16 + v + hf * 8;
        if (m < 60) FF[(long)m * PTOT + (long)b * NHWB + c0 + nt * 16 + coln] = (__bf16)acc[v];
    }
}

// ====================================================================
// K16: gather cat -> LayerNorm(180) -> WMMA MLP (180->60 relu ->60) -> +x
// grid 28800 (32 positions each), block 256 (8 waves).
// ====================================================================
__global__ void k_final(const __bf16* __restrict__ SF, const __bf16* __restrict__ AF,
                        const __bf16* __restrict__ FF, const float* __restrict__ gamma,
                        const float* __restrict__ beta, const float* __restrict__ w1g,
                        const float* __restrict__ w2g, const float* __restrict__ x,
                        float* __restrict__ out) {
    __shared__ __bf16 sw1[64 * 192];   // [oc][k]
    __shared__ __bf16 sw2[64 * 64];    // [oc][k]
    __shared__ __bf16 tc[192 * 32];    // [k][pos]
    __shared__ __bf16 hd[64 * 32];     // [k][pos]
    __shared__ float red1[8][32], red2[8][32], smu[32], sinv[32];
    int t = threadIdx.x;
    long p0 = (long)blockIdx.x * 32;
    int b = (int)(p0 / NHWB);
    int rem = (int)(p0 % NHWB);
    int n = rem / HW, hw0 = rem % HW;

    for (int idx = t; idx < 64 * 192; idx += 256) {
        int r = idx / 192, k = idx % 192;
        sw1[idx] = (__bf16)((r < 60 && k < 180) ? w1g[r * 180 + k] : 0.f);
    }
    for (int idx = t; idx < 64 * 64; idx += 256) {
        int r = idx >> 6, k = idx & 63;
        sw2[idx] = (__bf16)((r < 60 && k < 60) ? w2g[r * 60 + k] : 0.f);
    }
    long sfb = (long)b * HW * DNDIM;
    for (int idx = t; idx < 192 * 32; idx += 256) {
        int ch = idx >> 5, pos = idx & 31;
        float v = 0.f;
        if (ch < 60)       v = (float)SF[sfb + (long)(hw0 + pos) * DNDIM + ch * NANG + n];
        else if (ch < 120) v = (float)AF[(long)(ch - 60) * PTOT + p0 + pos];
        else if (ch < 180) v = (float)FF[(long)(ch - 120) * PTOT + p0 + pos];
        tc[idx] = (__bf16)v;
    }
    __syncthreads();
    // LayerNorm stats (rows 180..191 are zero and excluded by /180)
    int sub = t >> 5, pos = t & 31;
    float s1 = 0.f, s2 = 0.f;
    for (int r = 0; r < 24; ++r) {
        float v = (float)tc[(sub * 24 + r) * 32 + pos];
        s1 += v; s2 += v * v;
    }
    red1[sub][pos] = s1; red2[sub][pos] = s2; __syncthreads();
    if (t < 32) {
        float a = 0.f, c = 0.f;
        for (int s = 0; s < 8; ++s) { a += red1[s][t]; c += red2[s][t]; }
        float mu = a / 180.f;
        smu[t] = mu;
        sinv[t] = rsqrtf(c / 180.f - mu * mu + 1e-5f);
    }
    __syncthreads();
    for (int idx = t; idx < 192 * 32; idx += 256) {
        int ch = idx >> 5, pp = idx & 31;
        float v = (float)tc[idx];
        v = (ch < 180) ? ((v - smu[pp]) * sinv[pp] * gamma[ch] + beta[ch]) : 0.f;
        tc[idx] = (__bf16)v;
    }
    __syncthreads();
    // GEMM1: 60(pad64) x 32, K=192
    int wave = t >> 5, lane = t & 31;
    int mt = wave & 3, ct = wave >> 2;
    int coln = lane & 15, hf = lane >> 4;
    v8f acc = {};
#pragma unroll
    for (int ks = 0; ks < 6; ++ks) {
        v16bf a = load_frag(sw1 + ks * 32, 192, 1, mt * 16, lane);
        v16bf f = load_frag(tc + ks * 1024, 1, 32, ct * 16, lane);
        acc = wmma_bf16(a, f, acc);
    }
#pragma unroll
    for (int v = 0; v < 8; ++v) {
        int r = mt * 16 + v + hf * 8;
        hd[r * 32 + ct * 16 + coln] = (__bf16)fmaxf(acc[v], 0.f);   // relu; pad rows => 0
    }
    __syncthreads();
    // GEMM2: 60(pad64) x 32, K=64 ; + residual
    v8f acc2 = {};
#pragma unroll
    for (int ks = 0; ks < 2; ++ks) {
        v16bf a = load_frag(sw2 + ks * 32, 64, 1, mt * 16, lane);
        v16bf f = load_frag(hd + ks * 1024, 1, 32, ct * 16, lane);
        acc2 = wmma_bf16(a, f, acc2);
    }
#pragma unroll
    for (int v = 0; v < 8; ++v) {
        int c = mt * 16 + v + hf * 8;
        if (c < 60) {
            int pp = ct * 16 + coln;
            long xi = (((long)b * CC + c) * NANG + n) * HW + hw0 + pp;
            out[xi] = acc2[v] + x[xi];
        }
    }
}

// ====================================================================
extern "C" void kernel_launch(void* const* d_in, const int* in_sizes, int n_in,
                              void* d_out, int out_size, void* d_ws, size_t ws_size,
                              hipStream_t stream) {
    const float* x      = (const float*)d_in[0];
    const float* wsq    = (const float*)d_in[1];
    const float* wskv   = (const float*)d_in[2];
    const float* waqkv  = (const float*)d_in[3];
    const float* wfqkv  = (const float*)d_in[4];
    const float* gamma  = (const float*)d_in[5];
    const float* beta   = (const float*)d_in[6];
    const float* wm1    = (const float*)d_in[7];
    const float* wm2    = (const float*)d_in[8];
    float* out = (float*)d_out;

    char* ws = (char*)d_ws;
    auto alloc = [&](size_t bytes) { void* p = ws; ws += (bytes + 255) & ~(size_t)255; return p; };
    __bf16* PQ   = (__bf16*)alloc((size_t)60  * PTOT * 2);
    __bf16* PA   = (__bf16*)alloc((size_t)180 * PTOT * 2);
    __bf16* PF   = (__bf16*)alloc((size_t)180 * PTOT * 2);
    __bf16* PKV  = (__bf16*)alloc((size_t)120 * PLOW * 2);
    __bf16* S    = (__bf16*)alloc((size_t)BB * HW * H2W * 2);
    __bf16* SF   = (__bf16*)alloc((size_t)BB * HW * DNDIM * 2);
    __bf16* AF   = (__bf16*)alloc((size_t)60 * PTOT * 2);
    __bf16* FF   = (__bf16*)alloc((size_t)60 * PTOT * 2);
    float* rq_s  = (float*)alloc((size_t)BB * HW * 4);
    float* rk_s  = (float*)alloc((size_t)BB * H2W * 4);
    float* rqa   = (float*)alloc(100 * 4);
    float* rka   = (float*)alloc(100 * 4);
    float* rqf   = (float*)alloc(240 * 4);
    float* rkf   = (float*)alloc(240 * 4);
    float* Ga    = (float*)alloc(4 * 32 * 32 * 4);
    float* Gf    = (float*)alloc(4 * 64 * 64 * 4);
    __bf16* Aatt = (__bf16*)alloc(4 * 1024 * 2);
    __bf16* Fatt = (__bf16*)alloc(4 * 4096 * 2);

    hipMemsetAsync(Ga, 0, 4 * 32 * 32 * 4, stream);
    hipMemsetAsync(Gf, 0, 4 * 64 * 64 * 4, stream);

    k_proj_full<<<14400, 128, 0, stream>>>(x, wsq, waqkv, wfqkv, PQ, PA, PF);
    k_proj_low <<<3600, 128, 0, stream>>>(x, wskv, PKV);

    k_stat_sq  <<<144, 256, 0, stream>>>(PQ, rq_s);
    k_stat_sk  <<<36, 256, 0, stream>>>(PKV, rk_s);
    k_stat_ang <<<100, 256, 0, stream>>>(PA, rqa, rka);
    k_stat_fre <<<240, 256, 0, stream>>>(PF, rqf, rkf);

    k_qk_s     <<<dim3(36, 144, 4), 256, 0, stream>>>(PQ, PKV, S);
    k_softmax_s<<<BB * HW, 256, 0, stream>>>(S, rq_s, rk_s);

    k_gram_a   <<<dim3(270, 4), 128, 0, stream>>>(PA, Ga);
    k_gram_f   <<<dim3(225, 4), 256, 0, stream>>>(PF, Gf);
    k_softmax_a<<<1, 128, 0, stream>>>(Ga, rqa, rka, Aatt);
    k_softmax_f<<<1, 256, 0, stream>>>(Gf, rqf, rkf, Fatt);

    k_pv_s     <<<dim3(24, 144, 4), 256, 0, stream>>>(S, PKV, SF);
    k_pv_a     <<<dim3(8640, 4), 256, 0, stream>>>(Aatt, PA, AF);
    k_pv_f     <<<dim3(7200, 4), 256, 0, stream>>>(Fatt, PF, FF);

    k_final    <<<28800, 256, 0, stream>>>(SF, AF, FF, gamma, beta, wm1, wm2, x, out);
}
